// PolicyNet_43903155699922
// MI455X (gfx1250) — compile-verified, matrix operand order checked by reading, not run on hardware
//
#include <hip/hip_runtime.h>
#include <math.h>

// ---------------------------------------------------------------------------
// PolicyNet forward for MI455X (gfx1250, wave32, WMMA).
// GEMMs: f16 WMMA (v_wmma_f32_16x16x32_f16) with f32 accumulate. Tiles are
// staged into LDS with GLOBAL_LOAD_ASYNC_TO_LDS_B128 (ASYNCcnt) when shapes
// are tail-free; f32->f16 conversion happens on the LDS->fragment read side.
// Edge aggregation: f32 atomics (memory-bound; matrix units cannot help).
// ---------------------------------------------------------------------------

typedef __attribute__((ext_vector_type(16))) _Float16 v16h;
typedef __attribute__((ext_vector_type(8)))  float    v8f;

#if defined(__has_builtin)
#if __has_builtin(__builtin_amdgcn_global_load_async_to_lds_b128)
#define HAVE_ASYNC_LDS 1
#endif
#endif
#ifndef HAVE_ASYNC_LDS
#define HAVE_ASYNC_LDS 0
#endif

// exact parameter types for the async-LDS builtin:
// (v4i addrspace(1)*, v4i addrspace(3)*, imm offset, imm cpol)
typedef int vsi4 __attribute__((__vector_size__(16)));
typedef __attribute__((address_space(1))) vsi4 as1_vsi4;
typedef __attribute__((address_space(3))) vsi4 as3_vsi4;

// 16-byte global -> LDS copy (async on CDNA5; direct writes into LDS,
// no VGPR round trip, tracked with ASYNCcnt)
__device__ inline void async_copy16(const float* g, float* l) {
#if HAVE_ASYNC_LDS
  __builtin_amdgcn_global_load_async_to_lds_b128((as1_vsi4*)g, (as3_vsi4*)l, 0, 0);
#else
  *(float4*)l = *(const float4*)g;
#endif
}
__device__ inline void async_wait() {
#if HAVE_ASYNC_LDS
  asm volatile("s_wait_asynccnt 0x0" ::: "memory");
#else
  asm volatile("" ::: "memory");
#endif
}
__device__ inline void ds_reads_done() {
  asm volatile("s_wait_dscnt 0x0" ::: "memory");
}

// ---------------- block reduction helpers (128-thread blocks) --------------
__device__ inline float breduce_sum128(float v, float* red) {
  int t = threadIdx.x;
  red[t] = v; __syncthreads();
  #pragma unroll
  for (int s = 64; s > 0; s >>= 1) { if (t < s) red[t] += red[t + s]; __syncthreads(); }
  float r = red[0]; __syncthreads(); return r;
}
__device__ inline float breduce_max128(float v, float* red) {
  int t = threadIdx.x;
  red[t] = v; __syncthreads();
  #pragma unroll
  for (int s = 64; s > 0; s >>= 1) { if (t < s) red[t] = fmaxf(red[t], red[t + s]); __syncthreads(); }
  float r = red[0]; __syncthreads(); return r;
}

// ---------------- WMMA GEMM: C[M,N] (+bias/acc/relu) = A[M,K] @ B[K,N] ------
// BT=true : B stored [N,K] row-major (multiply by transpose of storage).
// TAIL=true : K not a multiple of 32 (guarded scalar staging, zero-pad).
// One wave per 16x16 tile; M and N must be multiples of 16 (true for all
// call sites: M in {50000,16384,1024}, N in {128,256,512,2048}).
template<bool BT, bool RELU, bool ACC, bool TAIL>
__global__ void __launch_bounds__(32)
wmma_gemm_kernel(const float* __restrict__ A, int lda,
                 const float* __restrict__ B, int ldb,
                 float* __restrict__ C, int ldc, int ccol0,
                 const float* __restrict__ bias,
                 int M, int N, int K) {
  __shared__ __align__(16) float lAf[16 * 32];   // [m][k]
  __shared__ __align__(16) float lBf[32 * 16];   // [k][n], or [n][k] when BT
  const int lane = threadIdx.x;
  const int tm = blockIdx.x * 16;
  const int tn = blockIdx.y * 16;
  v8f acc = {};

  for (int k0 = 0; k0 < K; k0 += 32) {
    if (!TAIL) {
      if (k0) ds_reads_done();           // prev tile's LDS reads retired
      // A tile 16x32 f32: 128 float4s, 4 async b128 ops per lane
      #pragma unroll
      for (int i = 0; i < 4; ++i) {
        int t = i * 32 + lane;
        int r = t >> 3, c4 = (t & 7) << 2;
        async_copy16(&A[(size_t)(tm + r) * lda + k0 + c4], &lAf[r * 32 + c4]);
      }
      // B tile 32x16 f32 (or 16x32 when BT)
      #pragma unroll
      for (int i = 0; i < 4; ++i) {
        int t = i * 32 + lane;
        if (BT) {
          int n = t >> 3, c4 = (t & 7) << 2;
          async_copy16(&B[(size_t)(tn + n) * ldb + k0 + c4], &lBf[n * 32 + c4]);
        } else {
          int kk = t >> 2, n4 = (t & 3) << 2;
          async_copy16(&B[(size_t)(k0 + kk) * ldb + tn + n4], &lBf[kk * 16 + n4]);
        }
      }
      if (k0 + 32 < K)                   // warm L2 for next chunk
        __builtin_prefetch(&A[(size_t)(tm + (lane >> 1)) * lda + k0 + 32], 0, 1);
      async_wait();
    } else {
      __syncthreads();
      #pragma unroll
      for (int i = 0; i < 16; ++i) {
        int idx = i * 32 + lane;
        int r = idx >> 5, c = idx & 31;
        lAf[idx] = (k0 + c < K) ? A[(size_t)(tm + r) * lda + k0 + c] : 0.f;
      }
      #pragma unroll
      for (int i = 0; i < 16; ++i) {
        int idx = i * 32 + lane;
        if (BT) {
          int n = idx >> 5, c = idx & 31;
          lBf[n * 32 + c] = (k0 + c < K) ? B[(size_t)(tn + n) * ldb + k0 + c] : 0.f;
        } else {
          int kk = idx >> 4, n = idx & 15;
          lBf[kk * 16 + n] = (k0 + kk < K) ? B[(size_t)(k0 + kk) * ldb + tn + n] : 0.f;
        }
      }
      __syncthreads();
    }

    // fragments per ISA 7.12.2 (16-bit A 16x32 layout; B mirrored),
    // f32->f16 conversion on the read side
    const int m  = lane & 15;            // row (A) / col (B)
    const int kg = (lane >> 4) << 3;     // 0 or 8
    v16h a, b;
    #pragma unroll
    for (int h = 0; h < 8; ++h) {
      a[h]     = (_Float16)lAf[m * 32 + (kg + h)];
      a[h + 8] = (_Float16)lAf[m * 32 + (16 + kg + h)];
      if (BT) {
        b[h]     = (_Float16)lBf[m * 32 + (kg + h)];
        b[h + 8] = (_Float16)lBf[m * 32 + (16 + kg + h)];
      } else {
        b[h]     = (_Float16)lBf[(kg + h) * 16 + m];
        b[h + 8] = (_Float16)lBf[(16 + kg + h) * 16 + m];
      }
    }
    acc = __builtin_amdgcn_wmma_f32_16x16x32_f16(false, a, false, b,
                                                 (short)0, acc, false, false);
  }

  // epilogue: C/D layout — element r: M = r + 8*(lane/16), N = lane%16
  const int n   = lane & 15;
  const int mrb = (lane >> 4) << 3;
  #pragma unroll
  for (int r = 0; r < 8; ++r) {
    int row = tm + mrb + r;
    int col = tn + n;
    size_t o = (size_t)row * ldc + ccol0 + col;
    float v = acc[r];
    if (bias) v += bias[col];
    if (ACC)  v += C[o];
    if (RELU) v = fmaxf(v, 0.f);
    C[o] = v;
  }
}

// ---------------- row normalization kernels --------------------------------
// MODE: 1 = LN+ReLU, 2 = LN+Softmax, 3 = L2norm+ReLU, 4 = Softmax
template<int MODE>
__global__ void __launch_bounds__(128)
row_norm_kernel(const float* __restrict__ in, float* __restrict__ out,
                const float* __restrict__ gamma, const float* __restrict__ beta,
                int D) {
  __shared__ float red[128];
  const int tid = threadIdx.x;
  const size_t row = blockIdx.x;
  const float* rin = in + row * (size_t)D;
  float* rout = out + row * (size_t)D;

  float buf[16];                 // D <= 2048, 128 threads
  float sum = 0.f, sumsq = 0.f, mx = -3.0e38f;
  int cnt = 0;
  for (int j = tid; j < D; j += 128) {
    float v = rin[j];
    buf[cnt++] = v;
    sum += v; sumsq += v * v; mx = fmaxf(mx, v);
  }

  if (MODE == 1 || MODE == 2) {
    float s  = breduce_sum128(sum, red);
    float sq = breduce_sum128(sumsq, red);
    float mean = s / (float)D;
    float var  = sq / (float)D - mean * mean;
    float inv  = rsqrtf(var + 1e-5f);
    cnt = 0;
    for (int j = tid; j < D; j += 128) {
      float y = (buf[cnt] - mean) * inv * gamma[j] + beta[j];
      buf[cnt++] = y;
    }
    if (MODE == 1) {
      cnt = 0;
      for (int j = tid; j < D; j += 128) { rout[j] = fmaxf(buf[cnt], 0.f); ++cnt; }
      return;
    }
    float lm = -3.0e38f; cnt = 0;
    for (int j = tid; j < D; j += 128) { lm = fmaxf(lm, buf[cnt]); ++cnt; }
    float gm = breduce_max128(lm, red);
    float ls = 0.f; cnt = 0;
    for (int j = tid; j < D; j += 128) { float e = __expf(buf[cnt] - gm); buf[cnt] = e; ls += e; ++cnt; }
    float gs = breduce_sum128(ls, red);
    cnt = 0;
    for (int j = tid; j < D; j += 128) { rout[j] = buf[cnt] / gs; ++cnt; }
    return;
  }
  if (MODE == 3) {               // l2 normalize then relu
    float sq = breduce_sum128(sumsq, red);
    float scale = 1.f / fmaxf(sqrtf(sq), 1e-12f);
    cnt = 0;
    for (int j = tid; j < D; j += 128) { rout[j] = fmaxf(buf[cnt] * scale, 0.f); ++cnt; }
    return;
  }
  if (MODE == 4) {               // plain softmax
    float gm = breduce_max128(mx, red);
    float ls = 0.f; cnt = 0;
    for (int j = tid; j < D; j += 128) { float e = __expf(buf[cnt] - gm); buf[cnt] = e; ls += e; ++cnt; }
    float gs = breduce_sum128(ls, red);
    cnt = 0;
    for (int j = tid; j < D; j += 128) { rout[j] = buf[cnt] / gs; ++cnt; }
  }
}

// ---------------- misc elementwise -----------------------------------------
__global__ void fill_kernel(float* p, float v, size_t n) {
  size_t i = (size_t)blockIdx.x * blockDim.x + threadIdx.x;
  size_t stride = (size_t)gridDim.x * blockDim.x;
  for (; i < n; i += stride) p[i] = v;
}

__global__ void agg_div_kernel(const float* num, const float* den, float* agg, size_t n) {
  size_t i = (size_t)blockIdx.x * blockDim.x + threadIdx.x;
  size_t stride = (size_t)gridDim.x * blockDim.x;
  for (; i < n; i += stride) {
    float d = den[i];
    agg[i] = (d > 0.f) ? num[i] / d : 0.f;
  }
}

// ---------------- edge (segment softmax) kernels ---------------------------
__device__ inline void atomicMaxFloat(float* addr, float val) {
  if (val >= 0.f) atomicMax((int*)addr, __float_as_int(val));
  else            atomicMin((unsigned int*)addr, (unsigned int)__float_as_int(val));
}

__global__ void __launch_bounds__(128)
edge_max_kernel(const int* __restrict__ src, const int* __restrict__ dst,
                const float* __restrict__ xp, const float* __restrict__ t,
                float* __restrict__ smax, int E, int C) {
  int e = blockIdx.x;
  int s = src[e], d = dst[e];
  for (int c = threadIdx.x; c < C; c += 128) {
    float val = xp[(size_t)s * C + c] * t[c];
    atomicMaxFloat(&smax[(size_t)d * C + c], val);
  }
}

__global__ void __launch_bounds__(128)
edge_sum_kernel(const int* __restrict__ src, const int* __restrict__ dst,
                const float* __restrict__ xp, const float* __restrict__ t,
                const float* __restrict__ smax,
                float* __restrict__ den, float* __restrict__ num, int E, int C) {
  int e = blockIdx.x;
  int s = src[e], d = dst[e];
  for (int c = threadIdx.x; c < C; c += 128) {
    float msg = xp[(size_t)s * C + c];
    float ex = __expf(msg * t[c] - smax[(size_t)d * C + c]);
    atomicAdd(&den[(size_t)d * C + c], ex);
    atomicAdd(&num[(size_t)d * C + c], ex * msg);
  }
}

// ---------------- MemPooling -----------------------------------------------
__global__ void __launch_bounds__(128)
mempool_assign_kernel(const float* __restrict__ h2, const float* __restrict__ keys,
                      const float* __restrict__ conv_w, float* __restrict__ S, int Nn) {
  __shared__ float lk[16 * 128];
  __shared__ float lcw[4];
  int tid = threadIdx.x;
  for (int i = tid; i < 16 * 128; i += 128) lk[i] = keys[i];
  if (tid < 4) lcw[tid] = conv_w[tid];
  __syncthreads();
  int n = blockIdx.x * 128 + tid;
  if (n >= Nn) return;
  float d2[16];
  #pragma unroll
  for (int i = 0; i < 16; ++i) d2[i] = 0.f;
  const float* xr = h2 + (size_t)n * 128;
  for (int f = 0; f < 128; ++f) {
    float xv = xr[f];
    #pragma unroll
    for (int i = 0; i < 16; ++i) { float df = lk[i * 128 + f] - xv; d2[i] += df * df; }
  }
  float spre[4] = {0.f, 0.f, 0.f, 0.f};
  for (int h = 0; h < 4; ++h) {
    float dist[4], ds = 0.f;
    #pragma unroll
    for (int kk = 0; kk < 4; ++kk) { dist[kk] = 1.f / (1.f + fmaxf(d2[h * 4 + kk], 0.f)); ds += dist[kk]; }
    #pragma unroll
    for (int kk = 0; kk < 4; ++kk) spre[kk] += lcw[h] * dist[kk] / ds;
  }
  float mx = fmaxf(fmaxf(spre[0], spre[1]), fmaxf(spre[2], spre[3]));
  float e[4], es = 0.f;
  #pragma unroll
  for (int kk = 0; kk < 4; ++kk) { e[kk] = __expf(spre[kk] - mx); es += e[kk]; }
  #pragma unroll
  for (int kk = 0; kk < 4; ++kk) S[(size_t)n * 4 + kk] = e[kk] / es;
}

__global__ void __launch_bounds__(256)
pool_reduce_kernel(const float* __restrict__ S, const float* __restrict__ h2,
                   float* __restrict__ pooled, int Nn) {
  __shared__ float red[256];
  int kk = blockIdx.x, f = blockIdx.y;
  float acc = 0.f;
  for (int n = threadIdx.x; n < Nn; n += 256)
    acc += S[(size_t)n * 4 + kk] * h2[(size_t)n * 128 + f];
  red[threadIdx.x] = acc; __syncthreads();
  for (int s = 128; s > 0; s >>= 1) {
    if (threadIdx.x < s) red[threadIdx.x] += red[threadIdx.x + s];
    __syncthreads();
  }
  if (threadIdx.x == 0) pooled[kk * 128 + f] = red[0];
}

// pooled[4,128] -> lin -> mean -> fx -> LN -> relu -> z[128]
__global__ void __launch_bounds__(128)
z_kernel(const float* __restrict__ pooled,
         const float* __restrict__ lin_w, const float* __restrict__ lin_b,
         const float* __restrict__ fx_w, const float* __restrict__ fx_b,
         const float* __restrict__ nx_g, const float* __restrict__ nx_b,
         float* __restrict__ z) {
  __shared__ float gg[128];
  __shared__ float red[128];
  int f = threadIdx.x;
  float acc = 0.f;
  for (int c = 0; c < 128; ++c) {
    float ps = pooled[c] + pooled[128 + c] + pooled[256 + c] + pooled[384 + c];
    acc += 0.25f * ps * lin_w[c * 128 + f];
  }
  acc += lin_b[f];
  gg[f] = acc; __syncthreads();
  float gz = fx_b[f];
  for (int c = 0; c < 128; ++c) gz += gg[c] * fx_w[c * 128 + f];
  float s  = breduce_sum128(gz, red);
  float sq = breduce_sum128(gz * gz, red);
  float mean = s / 128.f;
  float var  = sq / 128.f - mean * mean;
  float inv  = rsqrtf(var + 1e-5f);
  z[f] = fmaxf((gz - mean) * inv * nx_g[f] + nx_b[f], 0.f);
}

// ---------------- batch assembly -------------------------------------------
__global__ void __launch_bounds__(128)
hcat_kernel(const float* __restrict__ x, const float* __restrict__ emb,
            const float* __restrict__ log_return, const int* __restrict__ inp_types,
            const int* __restrict__ indexp, const float* __restrict__ z,
            float* __restrict__ hcat) {
  int b = blockIdx.x;
  int idx0 = indexp[0];
  int tp = inp_types[b];
  for (int j = threadIdx.x; j < 321; j += 128) {
    float v;
    if (j < 128)       v = x[(size_t)idx0 * 128 + j];
    else if (j < 192)  v = emb[tp * 64 + (j - 128)];
    else if (j == 192) v = log_return[b];
    else               v = z[j - 193];
    hcat[(size_t)b * 321 + j] = v;
  }
}

__global__ void __launch_bounds__(128)
concat2_kernel(const float* __restrict__ a, const float* __restrict__ b,
               float* __restrict__ out, int Da, int Db) {
  int r = blockIdx.x;
  int D = Da + Db;
  for (int j = threadIdx.x; j < D; j += 128)
    out[(size_t)r * D + j] = (j < Da) ? a[(size_t)r * Da + j]
                                      : b[(size_t)r * Db + (j - Da)];
}

__global__ void __launch_bounds__(64)
f3_kernel(const float* __restrict__ t2, const float* __restrict__ w,
          const float* __restrict__ b, float* __restrict__ out) {
  int r = blockIdx.x, o = threadIdx.x;
  if (o < 3) {
    float acc = b[o];
    for (int c = 0; c < 256; ++c) acc += t2[(size_t)r * 256 + c] * w[c * 3 + o];
    out[(size_t)r * 3 + o] = acc;
  }
}

// ---------------------------------------------------------------------------
extern "C" void kernel_launch(void* const* d_in, const int* in_sizes, int n_in,
                              void* d_out, int out_size, void* d_ws, size_t ws_size,
                              hipStream_t stream) {
  (void)n_in; (void)out_size; (void)ws_size;
  // Inputs flattened depth-first in setup_inputs() dict-insertion order:
  int p = 0;
  auto F = [&](int i) { return (const float*)d_in[i]; };
  const float* X      = F(p++);                    // x [N,128]
  const float* LOGRET = F(p++);                    // log_return [B]
  const float* EMB    = F(p++);                    // params.emb [8,64]
  const float* C1_PW = F(p++); const float* C1_PB = F(p++); const float* C1_T = F(p++);
  const float* C1_LLW = F(p++); const float* C1_LLB = F(p++); const float* C1_LRW = F(p++);
  const float* C2_PW = F(p++); const float* C2_PB = F(p++); const float* C2_T = F(p++);
  const float* C2_LLW = F(p++); const float* C2_LLB = F(p++); const float* C2_LRW = F(p++);
  const float* PX_K = F(p++); const float* PX_CW = F(p++);
  const float* PX_LW = F(p++); const float* PX_LB = F(p++);
  const float* GFX_W = F(p++); const float* GFX_B = F(p++);
  const float* GNX_G = F(p++); const float* GNX_B = F(p++);
  const float* FK_W = F(p++); const float* FK_B = F(p++);
  const float* FV_W = F(p++); const float* FV_B = F(p++);
  const float* NK_G = F(p++); const float* NK_B = F(p++);
  const float* NV_G = F(p++); const float* NV_B = F(p++);
  const float* MEMM = F(p++);                      // mem [8,2048,128]
  const float* M_FK_W = F(p++); const float* M_FK_B = F(p++);
  const float* M_FV_W = F(p++); const float* M_FV_B = F(p++);
  const float* M_FX_W = F(p++); const float* M_FX_B = F(p++);
  const float* M_NK_G = F(p++); const float* M_NK_B = F(p++);
  const float* M_NV_G = F(p++); const float* M_NV_B = F(p++);
  const float* M_NX_G = F(p++); const float* M_NX_B = F(p++);
  const float* F1_W = F(p++); const float* F1_B = F(p++);
  const float* N1_G = F(p++); const float* N1_B = F(p++);
  const float* F2_W = F(p++); const float* F2_B = F(p++);
  const float* N2_G = F(p++); const float* N2_B = F(p++);
  const float* F3_W = F(p++); const float* F3_B = F(p++);
  const int* EDGE   = (const int*)d_in[p++];       // [2,E]
  const int* ITYPES = (const int*)d_in[p++];       // [B]
  const int* INDEXP = (const int*)d_in[p++];       // scalar

  const int N = in_sizes[0] / 128;                 // 50000
  const int E = in_sizes[54] / 2;                  // 800000
  const int B = in_sizes[55];                      // 1024
  const int* SRC = EDGE;
  const int* DST = EDGE + E;

  // workspace layout (floats)
  float* w = (float*)d_ws;
  auto alloc = [&](size_t nfl) { float* q = w; w += nfl; return q; };
  float* P   = alloc((size_t)N * 256);
  float* MX  = alloc((size_t)N * 256);
  float* DN  = alloc((size_t)N * 256);
  float* NM  = alloc((size_t)N * 256);
  float* H1  = alloc((size_t)N * 256);
  float* H2  = alloc((size_t)N * 128);
  float* S4  = alloc((size_t)N * 4);
  float* POOL = alloc(4 * 128);
  float* Z   = alloc(128);
  float* KM  = alloc((size_t)16384 * 128);
  float* VM  = alloc((size_t)16384 * 128);
  float* HC  = alloc((size_t)B * 321);
  float* KQ  = alloc((size_t)B * 128);
  float* VQ  = alloc((size_t)B * 128);
  float* SC  = alloc((size_t)B * 2048);
  float* VV  = alloc((size_t)B * 1024);
  float* MO  = alloc((size_t)B * 128);
  float* HC2 = alloc((size_t)B * 256);
  float* T1  = alloc((size_t)B * 512);
  float* T2  = alloc((size_t)B * 256);

  dim3 w32(32), t128(128);

  // ---------------- SAGE layer 1 (128 -> 256) ----------------
  wmma_gemm_kernel<false, true, false, false><<<dim3(N/16, 8), w32, 0, stream>>>(
      X, 128, C1_PW, 128, P, 128, 0, C1_PB, N, 128, 128);
  fill_kernel<<<2048, 256, 0, stream>>>(MX, -INFINITY, (size_t)N * 128);
  fill_kernel<<<2048, 256, 0, stream>>>(DN, 0.f, (size_t)N * 128);
  fill_kernel<<<2048, 256, 0, stream>>>(NM, 0.f, (size_t)N * 128);
  edge_max_kernel<<<E, t128, 0, stream>>>(SRC, DST, P, C1_T, MX, E, 128);
  edge_sum_kernel<<<E, t128, 0, stream>>>(SRC, DST, P, C1_T, MX, DN, NM, E, 128);
  agg_div_kernel<<<2048, 256, 0, stream>>>(NM, DN, NM, (size_t)N * 128);
  wmma_gemm_kernel<false, false, false, false><<<dim3(N/16, 16), w32, 0, stream>>>(
      NM, 128, C1_LLW, 256, H1, 256, 0, C1_LLB, N, 256, 128);
  wmma_gemm_kernel<false, false, true, false><<<dim3(N/16, 16), w32, 0, stream>>>(
      X, 128, C1_LRW, 256, H1, 256, 0, nullptr, N, 256, 128);
  row_norm_kernel<3><<<N, t128, 0, stream>>>(H1, H1, nullptr, nullptr, 256);

  // ---------------- SAGE layer 2 (256 -> 128) ----------------
  wmma_gemm_kernel<false, true, false, false><<<dim3(N/16, 16), w32, 0, stream>>>(
      H1, 256, C2_PW, 256, P, 256, 0, C2_PB, N, 256, 256);
  fill_kernel<<<2048, 256, 0, stream>>>(MX, -INFINITY, (size_t)N * 256);
  fill_kernel<<<2048, 256, 0, stream>>>(DN, 0.f, (size_t)N * 256);
  fill_kernel<<<2048, 256, 0, stream>>>(NM, 0.f, (size_t)N * 256);
  edge_max_kernel<<<E, t128, 0, stream>>>(SRC, DST, P, C2_T, MX, E, 256);
  edge_sum_kernel<<<E, t128, 0, stream>>>(SRC, DST, P, C2_T, MX, DN, NM, E, 256);
  agg_div_kernel<<<2048, 256, 0, stream>>>(NM, DN, NM, (size_t)N * 256);
  wmma_gemm_kernel<false, false, false, false><<<dim3(N/16, 8), w32, 0, stream>>>(
      NM, 256, C2_LLW, 128, H2, 128, 0, C2_LLB, N, 128, 256);
  wmma_gemm_kernel<false, false, true, false><<<dim3(N/16, 8), w32, 0, stream>>>(
      H1, 256, C2_LRW, 128, H2, 128, 0, nullptr, N, 128, 256);
  row_norm_kernel<3><<<N, t128, 0, stream>>>(H2, H2, nullptr, nullptr, 128);

  // ---------------- MemPooling + global feature z ----------------
  mempool_assign_kernel<<<(N + 127) / 128, t128, 0, stream>>>(H2, PX_K, PX_CW, S4, N);
  pool_reduce_kernel<<<dim3(4, 128), 256, 0, stream>>>(S4, H2, POOL, N);
  z_kernel<<<1, t128, 0, stream>>>(POOL, PX_LW, PX_LB, GFX_W, GFX_B, GNX_G, GNX_B, Z);

  // ---------------- batch features, k/v (K=321 -> TAIL path) ----------------
  hcat_kernel<<<B, t128, 0, stream>>>(X, EMB, LOGRET, ITYPES, INDEXP, Z, HC);
  wmma_gemm_kernel<false, false, false, true><<<dim3(B/16, 8), w32, 0, stream>>>(
      HC, 321, FK_W, 128, KQ, 128, 0, FK_B, B, 128, 321);
  row_norm_kernel<2><<<B, t128, 0, stream>>>(KQ, KQ, NK_G, NK_B, 128);
  wmma_gemm_kernel<false, false, false, true><<<dim3(B/16, 8), w32, 0, stream>>>(
      HC, 321, FV_W, 128, VQ, 128, 0, FV_B, B, 128, 321);
  row_norm_kernel<1><<<B, t128, 0, stream>>>(VQ, VQ, NV_G, NV_B, 128);

  // ---------------- memory bank k/v transforms ----------------
  wmma_gemm_kernel<false, false, false, false><<<dim3(16384/16, 8), w32, 0, stream>>>(
      MEMM, 128, M_FK_W, 128, KM, 128, 0, M_FK_B, 16384, 128, 128);
  row_norm_kernel<2><<<16384, t128, 0, stream>>>(KM, KM, M_NK_G, M_NK_B, 128);
  wmma_gemm_kernel<false, false, false, false><<<dim3(16384/16, 8), w32, 0, stream>>>(
      MEMM, 128, M_FV_W, 128, VM, 128, 0, M_FV_B, 16384, 128, 128);
  row_norm_kernel<1><<<16384, t128, 0, stream>>>(VM, VM, M_NV_G, M_NV_B, 128);

  // ---------------- multi-head memory attention (8 heads) ----------------
  for (int h = 0; h < 8; ++h) {
    const float* KMh = KM + (size_t)h * 2048 * 128;
    const float* VMh = VM + (size_t)h * 2048 * 128;
    wmma_gemm_kernel<true, false, false, false><<<dim3(B/16, 2048/16), w32, 0, stream>>>(
        KQ, 128, KMh, 128, SC, 2048, 0, nullptr, B, 2048, 128);
    row_norm_kernel<4><<<B, t128, 0, stream>>>(SC, SC, nullptr, nullptr, 2048);
    wmma_gemm_kernel<false, false, false, false><<<dim3(B/16, 8), w32, 0, stream>>>(
        SC, 2048, VMh, 128, VV, 1024, h * 128, nullptr, B, 128, 2048);
  }
  wmma_gemm_kernel<false, false, false, false><<<dim3(B/16, 8), w32, 0, stream>>>(
      VV, 1024, M_FX_W, 128, MO, 128, 0, M_FX_B, B, 128, 1024);
  row_norm_kernel<1><<<B, t128, 0, stream>>>(MO, MO, M_NX_G, M_NX_B, 128);

  // ---------------- MLP head ----------------
  concat2_kernel<<<B, t128, 0, stream>>>(VQ, MO, HC2, 128, 128);
  wmma_gemm_kernel<false, false, false, false><<<dim3(B/16, 32), w32, 0, stream>>>(
      HC2, 256, F1_W, 512, T1, 512, 0, F1_B, B, 512, 256);
  row_norm_kernel<1><<<B, t128, 0, stream>>>(T1, T1, N1_G, N1_B, 512);
  wmma_gemm_kernel<false, false, false, false><<<dim3(B/16, 16), w32, 0, stream>>>(
      T1, 512, F2_W, 256, T2, 256, 0, F2_B, B, 256, 512);
  row_norm_kernel<1><<<B, t128, 0, stream>>>(T2, T2, N2_G, N2_B, 256);
  f3_kernel<<<B, 64, 0, stream>>>(T2, F3_W, F3_B, (float*)d_out);
}
